// UnrolledLSTM_9165460210001
// MI455X (gfx1250) — matrix operand order repaired and checked by reading, not compiled, not run on hardware
//
#include <hip/hip_runtime.h>
#include <hip/hip_bf16.h>

// ---------------- problem constants ----------------
#define BB   128            // batch
#define TT   512            // timesteps
#define II   256            // input size
#define HH   512            // hidden size
#define KK   (II + HH)      // 768, GEMM K
#define NG   4              // gates (f, u, g, o)
#define KB   (KK / 32)      // 24 K-blocks of 32

// padded LDS row strides (halves): row stride % 64dw != 0 -> conflict-free
#define XROW 264            // 256 + 4dw pad (TDM pad_interval=6 (128dw), pad_amount=3 (4dw))
#define HROW 520            // 512 + 4dw pad (TDM pad_interval=7 (256dw), pad_amount=3 (4dw))
#define LDS_ZX_BYTES (BB * XROW * 2)               //  67,584
#define LDS_ZH_BYTES (BB * HROW * 2)               // 133,120
#define LDS_BYTES    (LDS_ZX_BYTES + LDS_ZH_BYTES) // 200,704 (gbuf 64KB overlays)

typedef _Float16 v16h __attribute__((ext_vector_type(16)));
typedef _Float16 v8h  __attribute__((ext_vector_type(8)));
typedef float    v8f  __attribute__((ext_vector_type(8)));
typedef unsigned v4u  __attribute__((ext_vector_type(4)));
typedef int      v8i  __attribute__((ext_vector_type(8)));
typedef int      v4i  __attribute__((ext_vector_type(4)));

#if defined(__has_builtin)
#  if __has_builtin(__builtin_amdgcn_tensor_load_to_lds)
#    define USE_TDM 1
#  endif
#endif
#ifndef USE_TDM
#  define USE_TDM 0
#endif

// ---------------- workspace layout (bytes) ----------------
#define OFF_W16  ((size_t)0)
#define OFF_X16  ((size_t)3145728)
#define OFF_H16A ((size_t)(OFF_X16 + 33554432))
#define OFF_H16B ((size_t)(OFF_H16A + 131072))
#define OFF_H32  ((size_t)(OFF_H16B + 131072))
#define OFF_C32  ((size_t)(OFF_H32 + 262144))

// ---------------- prep: fp32 weights -> packed f16 [gate][n][k] ----------------
__global__ void lstm_cvt_w(const float* __restrict__ Wf, const float* __restrict__ Wu,
                           const float* __restrict__ Wc, const float* __restrict__ Wo,
                           _Float16* __restrict__ W16) {
    const size_t GW = (size_t)HH * KK;
    size_t i = (size_t)blockIdx.x * blockDim.x + threadIdx.x;
    if (i >= GW) return;
    W16[0 * GW + i] = (_Float16)Wf[i];
    W16[1 * GW + i] = (_Float16)Wu[i];
    W16[2 * GW + i] = (_Float16)Wc[i];
    W16[3 * GW + i] = (_Float16)Wo[i];
}

__global__ void lstm_cvt_x(const float* __restrict__ x, _Float16* __restrict__ x16, size_t n) {
    size_t i = (size_t)blockIdx.x * blockDim.x + threadIdx.x;
    if (i < n) x16[i] = (_Float16)x[i];
}

__global__ void lstm_init(_Float16* __restrict__ h16, float* __restrict__ c32) {
    int i = blockIdx.x * blockDim.x + threadIdx.x;
    if (i < BB * HH) { h16[i] = (_Float16)0.0f; c32[i] = 0.0f; }
}

__device__ __forceinline__ float sigmoid_f(float x) { return 1.0f / (1.0f + __expf(-x)); }
__device__ __forceinline__ float tanh_f(float x)    { return 2.0f / (1.0f + __expf(-2.0f * x)) - 1.0f; }

typedef __attribute__((address_space(3))) char lds_char;
__device__ __forceinline__ unsigned lds_addr_of(void* p) {
    return (unsigned)(uintptr_t)(lds_char*)p;
}

#if USE_TDM
// Issue one 2D TDM descriptor: rows x row_len (f16), global row stride, LDS row pad.
__device__ __forceinline__ void tdm_load_2d(unsigned lds_byte_addr, const _Float16* gptr,
                                            unsigned row_len, unsigned rows,
                                            unsigned g_stride_elems,
                                            unsigned pad_interval_code, unsigned pad_amount_code) {
    unsigned long long ga = (unsigned long long)(uintptr_t)gptr;
    // ---- D# group 0: count=1 | lds_addr | global_addr | type=2 ----
    v4u g0 = { 1u,
               lds_byte_addr,
               (unsigned)(ga & 0xFFFFFFFFull),
               (unsigned)((ga >> 32) & 0x1FFFFFFull) | (2u << 30) };
    // ---- D# group 1: data_size=2B, pad_enable, dims/tiles/stride ----
    v8i g1 = { (int)((1u << 16) | (1u << 20) | (pad_interval_code << 22) | (pad_amount_code << 25)),
               (int)(row_len << 16),            // tensor_dim0[15:0] at bits[63:48]
               (int)(rows << 16),               // tensor_dim1[15:0] at bits[95:80]
               (int)(row_len << 16),            // tile_dim0 at bits[127:112]
               (int)rows,                       // tile_dim1 at bits[143:128]
               (int)g_stride_elems,             // tensor_dim0_stride[31:0]
               0, 0 };
    v4i gz = { 0, 0, 0, 0 };
#if __clang_major__ >= 23
    v8i gz8 = { 0, 0, 0, 0, 0, 0, 0, 0 };
    __builtin_amdgcn_tensor_load_to_lds(g0, g1, gz, gz, gz8, 0);
#else
    __builtin_amdgcn_tensor_load_to_lds(g0, g1, gz, gz, 0);
#endif
}
#endif

// ---------------- one recurrent timestep ----------------
// grid = 16 WGs (32 hidden cols each), block = 256 threads = 8 wave32s.
// wave w: gate g = w>>1, col-tile ct = w&1 -> 16 cols, 8 row-tiles (M=128).
__global__ __launch_bounds__(256)
void lstm_step(const _Float16* __restrict__ x16, const _Float16* __restrict__ W16,
               const float* __restrict__ bf, const float* __restrict__ bu,
               const float* __restrict__ bc, const float* __restrict__ bo,
               const _Float16* __restrict__ h_in, _Float16* __restrict__ h_out,
               float* __restrict__ h32, float* __restrict__ c32,
               float* __restrict__ out, int t)
{
    extern __shared__ char smem[];
    _Float16* zx = (_Float16*)smem;                       // [128][XROW] staged x_t
    _Float16* zh = (_Float16*)(smem + LDS_ZX_BYTES);      // [128][HROW] staged h
    float*    gbuf = (float*)smem;                        // [4][128][32] overlays zx/zh later

    const int tid  = threadIdx.x;
    const int lane = tid & 31;
    const int w    = tid >> 5;
    const int g    = w >> 1;                // gate 0..3
    const int ct   = w & 1;                 // col-tile within WG
    const int blk  = blockIdx.x;            // 0..15
    const int jloc = ct * 16;
    const int j0   = blk * 32 + jloc;       // global hidden col of tile base
    const int half = lane >> 4;
    const int ln   = lane & 15;

    // ---- stage z = [x_t | h] into LDS (TDM DMA, padded conflict-free rows) ----
#if USE_TDM
    if (w == 0) {
        tdm_load_2d(lds_addr_of(zx), x16 + (size_t)t * II, II, BB, (unsigned)(TT * II), 6u, 3u);
        tdm_load_2d(lds_addr_of(zh), h_in,                 HH, BB, (unsigned)HH,        7u, 3u);
        __builtin_amdgcn_s_wait_tensorcnt(0);
    }
#else
    for (int c = tid; c < BB * (II / 8); c += 256) {      // 16B chunks of x rows
        const int m = c >> 5, kc = (c & 31) * 8;
        *(v8h*)(zx + m * XROW + kc) = *(const v8h*)(x16 + ((size_t)m * TT + t) * II + kc);
    }
    for (int c = tid; c < BB * (HH / 8); c += 256) {      // 16B chunks of h rows
        const int m = c >> 6, kc = (c & 63) * 8;
        *(v8h*)(zh + m * HROW + kc) = *(const v8h*)(h_in + (size_t)m * HH + kc);
    }
#endif
    __syncthreads();

    const v8f vzero = {0.f,0.f,0.f,0.f,0.f,0.f,0.f,0.f};
    v8f acc[8];
#pragma unroll
    for (int r = 0; r < 8; ++r) acc[r] = vzero;

    // B-matrix (weights) row for this lane's output column; 32B-aligned rows.
    const _Float16* wrow = W16 + ((size_t)(g * HH + j0 + ln)) * KK;

#pragma unroll 4
    for (int kb = 0; kb < KB; ++kb) {
        const int kbase = kb * 32;

        // B fragment: lane ln holds col n=j0+ln, K = kbase + half*16 + [0..15].
        v16h bfrag = *(const v16h*)(wrow + kbase + half * 16);
        if (kb + 2 < KB)
            __builtin_prefetch((const void*)(wrow + kbase + 64 + half * 16), 0, 1);

        // A source from LDS (K-block never straddles the x/h boundary).
        const _Float16* abase;
        int astride;
        if (kbase < II) { abase = zx + kbase + half * 16;        astride = XROW; }
        else            { abase = zh + (kbase - II) + half * 16; astride = HROW; }

        // Software-pipelined A fragments: issue ds_loads for rt+1 before WMMA rt
        // so the wait before each WMMA is a partial dscnt, overlapping LDS with
        // the matrix pipe.
        const _Float16* asrc0 = abase + (size_t)ln * astride;
        v8h lo = *(const v8h*)(asrc0);           // ds_load_b128
        v8h hi = *(const v8h*)(asrc0 + 8);       // ds_load_b128
#pragma unroll
        for (int rt = 0; rt < 8; ++rt) {
            v8h lo_n = lo, hi_n = hi;
            if (rt < 7) {
                const _Float16* nsrc = abase + (size_t)((rt + 1) * 16 + ln) * astride;
                lo_n = *(const v8h*)(nsrc);
                hi_n = *(const v8h*)(nsrc + 8);
            }
            v16h afrag;
#pragma unroll
            for (int i = 0; i < 8; ++i) { afrag[i] = lo[i]; afrag[i + 8] = hi[i]; }
            acc[rt] = __builtin_amdgcn_wmma_f32_16x16x32_f16(
                false, afrag, false, bfrag, (short)0, acc[rt], false, false);
            lo = lo_n; hi = hi_n;
        }
    }

    // staging region is dead; gbuf overlays it -> barrier first
    __syncthreads();

    // D layout: lane ln = tile col, VGPR r = row (half*8 + r).
#pragma unroll
    for (int rt = 0; rt < 8; ++rt)
#pragma unroll
        for (int r = 0; r < 8; ++r)
            gbuf[((g * BB) + rt * 16 + half * 8 + r) * 32 + jloc + ln] = acc[rt][r];

    __syncthreads();

    // elementwise LSTM cell: each thread owns 16 (m, jj) cells
#pragma unroll
    for (int q = 0; q < 16; ++q) {
        const int idx = tid * 16 + q;
        const int m  = idx >> 5;
        const int jj = idx & 31;
        const int j  = blk * 32 + jj;

        const float fv = sigmoid_f(gbuf[((0 * BB) + m) * 32 + jj] + bf[j]);
        const float uv = sigmoid_f(gbuf[((1 * BB) + m) * 32 + jj] + bu[j]);
        const float gv = tanh_f   (gbuf[((2 * BB) + m) * 32 + jj] + bc[j]);
        const float ov = sigmoid_f(gbuf[((3 * BB) + m) * 32 + jj] + bo[j]);

        const size_t hc = (size_t)m * HH + j;
        const float c2 = fv * c32[hc] + uv * gv;
        const float h2 = ov * tanh_f(c2);

        c32[hc]   = c2;
        h32[hc]   = h2;
        h_out[hc] = (_Float16)h2;
        out[((size_t)t * BB + m) * HH + j] = h2;   // outputs[t,0,m,j]
    }
}

// ---------------- tail: final h, c -> output ----------------
__global__ void lstm_tail(const float* __restrict__ h32, const float* __restrict__ c32,
                          float* __restrict__ out_tail) {
    int i = blockIdx.x * blockDim.x + threadIdx.x;
    if (i < BB * HH) {
        out_tail[i]           = h32[i];
        out_tail[BB * HH + i] = c32[i];
    }
}

extern "C" void kernel_launch(void* const* d_in, const int* in_sizes, int n_in,
                              void* d_out, int out_size, void* d_ws, size_t ws_size,
                              hipStream_t stream) {
    const float* x  = (const float*)d_in[0];
    const float* Wf = (const float*)d_in[1];
    const float* bf = (const float*)d_in[2];
    const float* Wu = (const float*)d_in[3];
    const float* bu = (const float*)d_in[4];
    const float* Wc = (const float*)d_in[5];
    const float* bc = (const float*)d_in[6];
    const float* Wo = (const float*)d_in[7];
    const float* bo = (const float*)d_in[8];
    float* out = (float*)d_out;
    char*  ws  = (char*)d_ws;

    _Float16* W16  = (_Float16*)(ws + OFF_W16);
    _Float16* X16  = (_Float16*)(ws + OFF_X16);
    _Float16* H16A = (_Float16*)(ws + OFF_H16A);
    _Float16* H16B = (_Float16*)(ws + OFF_H16B);
    float*    H32  = (float*)(ws + OFF_H32);
    float*    C32  = (float*)(ws + OFF_C32);

    const size_t GW = (size_t)HH * KK;
    const size_t NX = (size_t)BB * TT * II;

    lstm_cvt_w<<<(int)((GW + 255) / 256), 256, 0, stream>>>(Wf, Wu, Wc, Wo, W16);
    lstm_cvt_x<<<(int)((NX + 255) / 256), 256, 0, stream>>>(x, X16, NX);
    lstm_init<<<(BB * HH + 255) / 256, 256, 0, stream>>>(H16A, C32);

    for (int t = 0; t < TT; ++t) {
        _Float16* hin  = (t & 1) ? H16B : H16A;
        _Float16* hout = (t & 1) ? H16A : H16B;
        lstm_step<<<HH / 32, 256, LDS_BYTES, stream>>>(X16, W16, bf, bu, bc, bo,
                                                       hin, hout, H32, C32, out, t);
    }

    lstm_tail<<<(BB * HH + 255) / 256, 256, 0, stream>>>(H32, C32,
                                                         out + (size_t)TT * BB * HH);
}
